// ARDecoder_46265387712738
// MI455X (gfx1250) — matrix operand order repaired and checked by reading, not compiled, not run on hardware
//
#include <hip/hip_runtime.h>
#include <hip/hip_bf16.h>
#include <stdint.h>

// ---------------------------------------------------------------------------
// Types
// ---------------------------------------------------------------------------
typedef __bf16 bf16;
typedef __attribute__((ext_vector_type(16))) __bf16 v16bf;
typedef __attribute__((ext_vector_type(8)))  __bf16 bf16x8;
typedef __attribute__((ext_vector_type(8)))  float  v8f;

// ---------------------------------------------------------------------------
// Problem dims (fixed per reference)
// ---------------------------------------------------------------------------
#define BB   2
#define TT   1024
#define DD   1024
#define HH   16
#define KVH  4
#define HDIM 64
#define DFFN 4096
#define VV   32000
#define PPD  256
#define DLATD 512
#define LLAY 4
#define MTOK (BB*TT)          // 2048 token rows

// ---------------------------------------------------------------------------
// CDNA5 async global->LDS (ASYNCcnt-tracked DMA, no VGPR round trip).
// vdst VGPR = LDS byte offset (low 32 bits of generic shared pointer),
// vaddr = 64-bit global address VGPR pair.
// ---------------------------------------------------------------------------
__device__ __forceinline__ void async_load_b128(uint32_t lds_off, const void* gaddr) {
  asm volatile("global_load_async_to_lds_b128 %0, %1, off"
               :: "v"(lds_off), "v"(gaddr)
               : "memory");
}

__device__ __forceinline__ void wait_async0() {
#if __has_builtin(__builtin_amdgcn_s_wait_asynccnt)
  __builtin_amdgcn_s_wait_asynccnt(0);
#else
  asm volatile("s_wait_asynccnt 0x0" ::: "memory");
#endif
}

__device__ __forceinline__ uint32_t lds_off_of(const void* p) {
  return (uint32_t)(uintptr_t)p;  // generic shared addr low 32 bits == DS offset
}

// ---------------------------------------------------------------------------
// f32 -> bf16 cast (grid-stride)
// ---------------------------------------------------------------------------
__global__ __launch_bounds__(256) void cast_f32_bf16(const float* __restrict__ src,
                                                     bf16* __restrict__ dst, long long n) {
  long long i = (long long)blockIdx.x * blockDim.x + threadIdx.x;
  long long stride = (long long)gridDim.x * blockDim.x;
  for (; i < n; i += stride) dst[i] = (bf16)src[i];
}

// ---------------------------------------------------------------------------
// z0 @ W_z0^T  -> zbias (B, D)   (tiny: 2x1024 outputs, K=512)
// ---------------------------------------------------------------------------
__global__ __launch_bounds__(256) void zproj_kernel(const float* __restrict__ z0,
                                                    const float* __restrict__ Wz,
                                                    float* __restrict__ zb) {
  int i = blockIdx.x * blockDim.x + threadIdx.x;
  if (i >= BB * DD) return;
  int b = i / DD, d = i % DD;
  float acc = 0.f;
  for (int k2 = 0; k2 < DLATD; k2++) acc += z0[b * DLATD + k2] * Wz[d * DLATD + k2];
  zb[i] = acc;
}

// ---------------------------------------------------------------------------
// Embedding gather: out[m, :] = E_bf16[x[m], :]
// ---------------------------------------------------------------------------
__global__ __launch_bounds__(256) void gather_embed(const int* __restrict__ x,
                                                    const bf16* __restrict__ Ebf,
                                                    bf16* __restrict__ out) {
  int i = blockIdx.x * blockDim.x + threadIdx.x;
  if (i >= MTOK * PPD) return;
  int m = i / PPD, p = i % PPD;
  out[i] = Ebf[(long long)x[m] * PPD + p];
}

// ---------------------------------------------------------------------------
// RMSNorm: one block per row. Writes bf16 always; f32 optionally.
// ---------------------------------------------------------------------------
__global__ __launch_bounds__(256) void rmsnorm_kernel(const float* __restrict__ in,
                                                      const float* __restrict__ w,
                                                      bf16* __restrict__ outb,
                                                      float* __restrict__ outf, int N) {
  int m = blockIdx.x;
  const float* row = in + (size_t)m * N;
  __shared__ float red[256];
  float s = 0.f;
  for (int i = threadIdx.x; i < N; i += 256) { float v = row[i]; s += v * v; }
  red[threadIdx.x] = s;
  __syncthreads();
  for (int off = 128; off > 0; off >>= 1) {
    if (threadIdx.x < off) red[threadIdx.x] += red[threadIdx.x + off];
    __syncthreads();
  }
  float scale = rsqrtf(red[0] / (float)N + 1e-6f);
  for (int i = threadIdx.x; i < N; i += 256) {
    float v = row[i] * scale * w[i];
    outb[(size_t)m * N + i] = (bf16)v;
    if (outf) outf[(size_t)m * N + i] = v;
  }
}

// ---------------------------------------------------------------------------
// act = silu(g) * u  -> bf16
// ---------------------------------------------------------------------------
__global__ __launch_bounds__(256) void silu_mul_kernel(const float* __restrict__ g,
                                                       const float* __restrict__ u,
                                                       bf16* __restrict__ out, long long n) {
  long long i = (long long)blockIdx.x * blockDim.x + threadIdx.x;
  long long stride = (long long)gridDim.x * blockDim.x;
  for (; i < n; i += stride) {
    float gg = g[i];
    float s = gg / (1.f + __expf(-gg));
    out[i] = (bf16)(s * u[i]);
  }
}

// ---------------------------------------------------------------------------
// WMMA GEMM: C(MxN) = A(MxK) * Bw(NxK)^T, bf16 in / f32 accumulate.
// Requires M%128==0, N%128==0, K%32==0 (true for all call sites).
// flags: bit0 = accumulate into existing C; bit1 = add bias[(row>>10)*N + col]
// Block: 256 thr (8 waves). Tile 128x128x32. Wave = 64x32 = 4x2 wmma tiles.
// Global->LDS staging uses CDNA5 async-to-LDS DMA with 2-stage double buffer:
// issue async loads for slab k+32, compute WMMAs on slab k, then
// s_wait_asynccnt 0 + barrier.
// ---------------------------------------------------------------------------
__global__ __launch_bounds__(256) void gemm_bf16_wmma(const bf16* __restrict__ A,
                                                      const bf16* __restrict__ Bw,
                                                      float* __restrict__ C,
                                                      const float* __restrict__ bias,
                                                      int M, int N, int K, int flags) {
  __shared__ __align__(16) bf16 As[2][128 * 40];
  __shared__ __align__(16) bf16 Bs[2][128 * 40];
  int tid = threadIdx.x;
  int lane = tid & 31, wid = tid >> 5;
  int warpM = wid >> 2, warpN = wid & 3;   // 2 x 4 wave grid
  int lm = lane & 15, hf = lane >> 4;
  int m0 = blockIdx.y * 128, n0 = blockIdx.x * 128;

  // this thread's two (row, col8) chunks within the 128x32 tile
  int rowc[2], colc[2];
  for (int part = 0; part < 2; part++) {
    int chunk = tid + part * 256;          // 0..511
    rowc[part] = chunk >> 2;
    colc[part] = (chunk & 3) * 8;
  }

  v8f acc[4][2];
  for (int i = 0; i < 4; i++)
    for (int j = 0; j < 2; j++)
      for (int r = 0; r < 8; r++) acc[i][j][r] = 0.f;

  // prologue: stage first K slab into buffer 0
  for (int part = 0; part < 2; part++) {
    int row = rowc[part], col8 = colc[part];
    async_load_b128(lds_off_of(&As[0][row * 40 + col8]),
                    &A[(size_t)(m0 + row) * K + col8]);
    async_load_b128(lds_off_of(&Bs[0][row * 40 + col8]),
                    &Bw[(size_t)(n0 + row) * K + col8]);
  }
  wait_async0();
  __syncthreads();

  int cur = 0;
  for (int k0 = 0; k0 < K; k0 += 32) {
    // prefetch next slab into the other buffer (async DMA, overlaps WMMAs)
    if (k0 + 32 < K) {
      int nxt = cur ^ 1;
      for (int part = 0; part < 2; part++) {
        int row = rowc[part], col8 = colc[part];
        async_load_b128(lds_off_of(&As[nxt][row * 40 + col8]),
                        &A[(size_t)(m0 + row) * K + k0 + 32 + col8]);
        async_load_b128(lds_off_of(&Bs[nxt][row * 40 + col8]),
                        &Bw[(size_t)(n0 + row) * K + k0 + 32 + col8]);
      }
    }

    // A fragments (16x32, M x K): lane half selects K subset {0-7,16-23} vs {8-15,24-31}
    v16bf afrag[4], bfrag[2];
    for (int i = 0; i < 4; i++) {
      int rowA = warpM * 64 + i * 16 + lm;
      int koff = hf * 8;
      bf16x8 lo = *(const bf16x8*)&As[cur][rowA * 40 + koff];
      bf16x8 hi2 = *(const bf16x8*)&As[cur][rowA * 40 + koff + 16];
      ((bf16x8*)&afrag[i])[0] = lo;
      ((bf16x8*)&afrag[i])[1] = hi2;
    }
    // B fragments (32x16, K x N): lane = column n; half selects K subset {0-7,16-23} vs {8-15,24-31}
    for (int j = 0; j < 2; j++) {
      int colB = warpN * 32 + j * 16 + lm;
      int koff = hf * 8;
      bf16x8 lo = *(const bf16x8*)&Bs[cur][colB * 40 + koff];
      bf16x8 hi2 = *(const bf16x8*)&Bs[cur][colB * 40 + koff + 16];
      ((bf16x8*)&bfrag[j])[0] = lo;
      ((bf16x8*)&bfrag[j])[1] = hi2;
    }
    for (int i = 0; i < 4; i++)
      for (int j = 0; j < 2; j++)
        acc[i][j] = __builtin_amdgcn_wmma_f32_16x16x32_bf16(
            false, afrag[i], false, bfrag[j], (short)0, acc[i][j], false, false);

    // own-wave async completion + block barrier => next buffer valid for all
    wait_async0();
    __syncthreads();
    cur ^= 1;
  }

  // epilogue: C layout = lane column n, VGPR r -> row r + 8*half
  for (int i = 0; i < 4; i++)
    for (int j = 0; j < 2; j++) {
      int colg = n0 + warpN * 32 + j * 16 + lm;
      for (int r = 0; r < 8; r++) {
        int rowg = m0 + warpM * 64 + i * 16 + r + hf * 8;
        size_t idx = (size_t)rowg * N + colg;
        float v = acc[i][j][r];
        if (flags & 2) v += bias[(rowg >> 10) * N + colg];  // T==1024
        if (flags & 1) C[idx] += v;
        else           C[idx] = v;
      }
    }
}

// ---------------------------------------------------------------------------
// Flash attention (causal, GQA). One wave per (b, h, 16-query tile).
// S^T = K * Q^T via WMMA (lane = query column -> in-lane softmax stats),
// then the S^T C-layout fragments ARE the A-layout of P, so P*V is a
// direct WMMA with only f32->bf16 packing in between.
// ---------------------------------------------------------------------------
__global__ __launch_bounds__(32) void attn_kernel(const float* __restrict__ q,
                                                  const float* __restrict__ k,
                                                  const float* __restrict__ v,
                                                  float* __restrict__ o) {
  int qb = blockIdx.x * 16;
  int bh = blockIdx.y;
  int b = bh / HH, h = bh % HH;
  int kvh = h / (HH / KVH);
  int lane = threadIdx.x;
  int qc = lane & 15, hf = lane >> 4;
  const float inv_sqrt_hd = 0.125f;  // 1/sqrt(64)

  // Q as B-operand of S^T (columns = queries): elements e -> hd = e + 16*half
  v16bf qf0, qf1;
  {
    size_t base = ((size_t)(b * TT + qb + qc) * HH + h) * HDIM;
    for (int e = 0; e < 16; e++) {
      qf0[e] = (bf16)q[base + hf * 16 + e];
      qf1[e] = (bf16)q[base + 32 + hf * 16 + e];
    }
  }

  v8f occ[4];
  for (int c = 0; c < 4; c++)
    for (int r = 0; r < 8; r++) occ[c][r] = 0.f;
  float m_run = -1e30f, l_run = 0.f;

  for (int kb = 0; kb <= qb + 15; kb += 32) {
    v8f s0, s1;
    for (int s = 0; s < 2; s++) {
      int ks = kb + 16 * s;
      // K rows as A-operand (16 keys x 64 hd, split into two k=32 wmmas)
      size_t kbase = ((size_t)(b * TT + ks + qc) * KVH + kvh) * HDIM;
      v16bf ka0, ka1;
      for (int e = 0; e < 16; e++) {
        int hd = e + 16 * hf;  // contiguous K order, same as Q
        ka0[e] = (bf16)k[kbase + hd];
        ka1[e] = (bf16)k[kbase + 32 + hd];
      }
      v8f cs;
      for (int r = 0; r < 8; r++) cs[r] = 0.f;
      cs = __builtin_amdgcn_wmma_f32_16x16x32_bf16(false, ka0, false, qf0, (short)0, cs, false, false);
      cs = __builtin_amdgcn_wmma_f32_16x16x32_bf16(false, ka1, false, qf1, (short)0, cs, false, false);
      if (s == 0) s0 = cs; else s1 = cs;
    }

    // scale + causal mask; in-lane stats over 16 keys, partner lane has other 16
    int qg = qb + qc;
    float mblk = -1e30f;
    for (int r = 0; r < 8; r++) {
      int key0 = kb + r + 8 * hf;
      int key1 = kb + 16 + r + 8 * hf;
      float v0 = s0[r] * inv_sqrt_hd; if (key0 > qg) v0 = -1e30f;
      float v1 = s1[r] * inv_sqrt_hd; if (key1 > qg) v1 = -1e30f;
      s0[r] = v0; s1[r] = v1;
      mblk = fmaxf(mblk, fmaxf(v0, v1));
    }
    mblk = fmaxf(mblk, __shfl_xor(mblk, 16, 32));
    float m_new = fmaxf(m_run, mblk);
    float alpha = __expf(m_run - m_new);
    float p0[8], p1[8];
    float ls = 0.f;
    for (int r = 0; r < 8; r++) {
      float e0 = (s0[r] < -1e29f) ? 0.f : __expf(s0[r] - m_new);
      float e1 = (s1[r] < -1e29f) ? 0.f : __expf(s1[r] - m_new);
      p0[r] = e0; p1[r] = e1; ls += e0 + e1;
    }
    ls += __shfl_xor(ls, 16, 32);
    l_run = l_run * alpha + ls;
    m_run = m_new;

    // rescale O accumulators (row q = r + 8*half; alpha lives in lane q)
    for (int r = 0; r < 8; r++) {
      float ar = __shfl(alpha, r + 8 * hf, 32);
      for (int c = 0; c < 4; c++) occ[c][r] *= ar;
    }

    // pack P (S^T C-layout == P A-layout): element e<8 from tile0, else tile1
    v16bf pf;
    for (int e = 0; e < 16; e++) pf[e] = (bf16)(e < 8 ? p0[e] : p1[e - 8]);

    // V as B-operand (32 keys x 16 hd per chunk), 4 chunks cover HD=64
    for (int c = 0; c < 4; c++) {
      v16bf vf;
      for (int e = 0; e < 16; e++) {
        int key = kb + 8 * hf + e + (e >= 8 ? 8 : 0);
        vf[e] = (bf16)v[((size_t)(b * TT + key) * KVH + kvh) * HDIM + c * 16 + qc];
      }
      occ[c] = __builtin_amdgcn_wmma_f32_16x16x32_bf16(false, pf, false, vf, (short)0, occ[c], false, false);
    }
  }

  // normalize and store
  for (int r = 0; r < 8; r++) {
    float lr = __shfl(l_run, r + 8 * hf, 32);
    float linv = 1.f / lr;
    int qg = qb + r + 8 * hf;
    for (int c = 0; c < 4; c++)
      o[((size_t)(b * TT + qg) * HH + h) * HDIM + c * 16 + qc] = occ[c][r] * linv;
  }
}

// ---------------------------------------------------------------------------
// Host-side orchestration
// ---------------------------------------------------------------------------
extern "C" void kernel_launch(void* const* d_in, const int* in_sizes, int n_in,
                              void* d_out, int out_size, void* d_ws, size_t ws_size,
                              hipStream_t stream) {
  (void)in_sizes; (void)n_in; (void)out_size; (void)ws_size;
  const int*   x      = (const int*)  d_in[0];
  const float* z0     = (const float*)d_in[1];
  const float* E      = (const float*)d_in[2];
  const float* Weu    = (const float*)d_in[3];
  const float* Wz     = (const float*)d_in[4];
  const float* patchW = (const float*)d_in[5];
  const float* fnw    = (const float*)d_in[6];
  const float* n1w    = (const float*)d_in[7];
  const float* qnw    = (const float*)d_in[8];
  const float* knw    = (const float*)d_in[9];
  const float* n2w    = (const float*)d_in[10];
  const float* Wq     = (const float*)d_in[11];
  const float* Wk     = (const float*)d_in[12];
  const float* Wv     = (const float*)d_in[13];
  const float* Wo     = (const float*)d_in[14];
  const float* Wg     = (const float*)d_in[15];
  const float* Wu     = (const float*)d_in[16];
  const float* Wd     = (const float*)d_in[17];
  float* logits = (float*)d_out;

  // bump allocator over workspace
  char* ws = (char*)d_ws;
  size_t off = 0;
  auto alloc = [&](size_t bytes) -> void* {
    void* p = ws + off;
    off += (bytes + 255) & ~(size_t)255;
    return p;
  };

  // bf16 weight copies
  bf16* Ebf     = (bf16*)alloc((size_t)VV * PPD * 2);
  bf16* Weubf   = (bf16*)alloc((size_t)DD * PPD * 2);
  bf16* patchbf = (bf16*)alloc((size_t)PPD * DD * 2);
  bf16* Wqbf    = (bf16*)alloc((size_t)LLAY * DD * DD * 2);
  bf16* Wkbf    = (bf16*)alloc((size_t)LLAY * (KVH * HDIM) * DD * 2);
  bf16* Wvbf    = (bf16*)alloc((size_t)LLAY * (KVH * HDIM) * DD * 2);
  bf16* Wobf    = (bf16*)alloc((size_t)LLAY * DD * DD * 2);
  bf16* Wgbf    = (bf16*)alloc((size_t)LLAY * DFFN * DD * 2);
  bf16* Wubf    = (bf16*)alloc((size_t)LLAY * DFFN * DD * 2);
  bf16* Wdbf    = (bf16*)alloc((size_t)LLAY * DD * DFFN * 2);

  // activations
  float* h     = (float*)alloc((size_t)MTOK * DD * 4);
  float* zbias = (float*)alloc((size_t)BB * DD * 4);
  bf16*  Egbf  = (bf16*) alloc((size_t)MTOK * PPD * 2);
  float* hnf   = (float*)alloc((size_t)MTOK * DD * 4);
  bf16*  hnbf  = (bf16*) alloc((size_t)MTOK * DD * 2);
  bf16*  qnbf  = (bf16*) alloc((size_t)MTOK * DD * 2);
  bf16*  knbf  = (bf16*) alloc((size_t)MTOK * DD * 2);
  float* qf    = (float*)alloc((size_t)MTOK * DD * 4);
  float* kf    = (float*)alloc((size_t)MTOK * KVH * HDIM * 4);
  float* vf    = (float*)alloc((size_t)MTOK * KVH * HDIM * 4);
  float* of    = (float*)alloc((size_t)MTOK * DD * 4);
  bf16*  obf   = (bf16*) alloc((size_t)MTOK * DD * 2);
  bf16*  fnbf  = (bf16*) alloc((size_t)MTOK * DD * 2);
  float* gfb   = (float*)alloc((size_t)MTOK * DFFN * 4);
  float* ufb   = (float*)alloc((size_t)MTOK * DFFN * 4);
  bf16*  actbf = (bf16*) alloc((size_t)MTOK * DFFN * 2);
  bf16*  hfbf  = (bf16*) alloc((size_t)MTOK * DD * 2);
  float* pf    = (float*)alloc((size_t)MTOK * PPD * 4);
  bf16*  pbf   = (bf16*) alloc((size_t)MTOK * PPD * 2);

  auto cast = [&](const float* s, bf16* d, long long n) {
    int blocks = (int)((n + 255) / 256);
    if (blocks > 4096) blocks = 4096;
    cast_f32_bf16<<<blocks, 256, 0, stream>>>(s, d, n);
  };
  auto gemm = [&](const bf16* A, const bf16* Bw, float* C, const float* bias,
                  int M, int N, int K, int flags) {
    dim3 g(N / 128, M / 128);
    gemm_bf16_wmma<<<g, 256, 0, stream>>>(A, Bw, C, bias, M, N, K, flags);
  };
  auto rms = [&](const float* in, const float* w, bf16* ob, float* ofp, int N) {
    rmsnorm_kernel<<<MTOK, 256, 0, stream>>>(in, w, ob, ofp, N);
  };

  // ---- weight casts ----
  cast(E,      Ebf,     (long long)VV * PPD);
  cast(Weu,    Weubf,   (long long)DD * PPD);
  cast(patchW, patchbf, (long long)PPD * DD);
  cast(Wq,     Wqbf,    (long long)LLAY * DD * DD);
  cast(Wk,     Wkbf,    (long long)LLAY * KVH * HDIM * DD);
  cast(Wv,     Wvbf,    (long long)LLAY * KVH * HDIM * DD);
  cast(Wo,     Wobf,    (long long)LLAY * DD * DD);
  cast(Wg,     Wgbf,    (long long)LLAY * DFFN * DD);
  cast(Wu,     Wubf,    (long long)LLAY * DFFN * DD);
  cast(Wd,     Wdbf,    (long long)LLAY * DD * DFFN);

  // ---- embed + latent conditioning ----
  zproj_kernel<<<(BB * DD + 255) / 256, 256, 0, stream>>>(z0, Wz, zbias);
  gather_embed<<<(MTOK * PPD + 255) / 256, 256, 0, stream>>>(x, Ebf, Egbf);
  gemm(Egbf, Weubf, h, zbias, MTOK, DD, PPD, /*bias_bt*/ 2);

  // ---- layers ----
  for (int i = 0; i < LLAY; i++) {
    rms(h, n1w + (size_t)i * DD, hnbf, hnf, DD);
    rms(hnf, qnw + (size_t)i * DD, qnbf, nullptr, DD);
    rms(hnf, knw + (size_t)i * DD, knbf, nullptr, DD);

    gemm(qnbf, Wqbf + (size_t)i * DD * DD, qf, nullptr, MTOK, DD, DD, 0);
    gemm(knbf, Wkbf + (size_t)i * KVH * HDIM * DD, kf, nullptr, MTOK, KVH * HDIM, DD, 0);
    gemm(hnbf, Wvbf + (size_t)i * KVH * HDIM * DD, vf, nullptr, MTOK, KVH * HDIM, DD, 0);

    attn_kernel<<<dim3(TT / 16, BB * HH), 32, 0, stream>>>(qf, kf, vf, of);
    cast(of, obf, (long long)MTOK * DD);
    gemm(obf, Wobf + (size_t)i * DD * DD, h, nullptr, MTOK, DD, DD, /*acc*/ 1);

    rms(h, n2w + (size_t)i * DD, fnbf, nullptr, DD);
    gemm(fnbf, Wgbf + (size_t)i * DFFN * DD, gfb, nullptr, MTOK, DFFN, DD, 0);
    gemm(fnbf, Wubf + (size_t)i * DFFN * DD, ufb, nullptr, MTOK, DFFN, DD, 0);
    silu_mul_kernel<<<4096, 256, 0, stream>>>(gfb, ufb, actbf, (long long)MTOK * DFFN);
    gemm(actbf, Wdbf + (size_t)i * DD * DFFN, h, nullptr, MTOK, DD, DFFN, /*acc*/ 1);
  }

  // ---- head: final norm -> patch proj -> tied-embedding logits ----
  rms(h, fnw, hfbf, nullptr, DD);
  gemm(hfbf, patchbf, pf, nullptr, MTOK, PPD, DD, 0);
  cast(pf, pbf, (long long)MTOK * PPD);
  gemm(pbf, Ebf, logits, nullptr, MTOK, VV, PPD, 0);
}
